// MultiHeadedAttention_2156073583112
// MI455X (gfx1250) — compile-verified, matrix operand order checked by reading, not run on hardware
//
#include <hip/hip_runtime.h>
#include <hip/hip_bf16.h>
#include <math.h>

// ---------------------------------------------------------------------------
// MultiHeadedAttention for MI455X (gfx1250, wave32, WMMA 16x16x32 f16->f32)
// B=2, S=2048, D=1024, H=16, d_k=64
// ---------------------------------------------------------------------------

typedef _Float16 half16 __attribute__((ext_vector_type(16)));
typedef _Float16 half8  __attribute__((ext_vector_type(8)));
typedef float    float8 __attribute__((ext_vector_type(8)));

#define BATCH 2
#define SEQ   2048
#define DMODEL 1024
#define NHEAD 16
#define DK    64

static __device__ __forceinline__ float8 wmma_f16(half16 a, half16 b, float8 c) {
    return __builtin_amdgcn_wmma_f32_16x16x32_f16(
        /*neg_a=*/false, a, /*neg_b=*/false, b,
        /*c_mod=*/(short)0, c, /*reuse_a=*/false, /*reuse_b=*/false);
}

// Load a 16x32 fragment (A-layout, and by mirror-symmetry B^T rows) from a
// row-major f16 matrix. Documented CDNA5 16-bit A layout:
//   lanes 0-15 : M=lane,    K in {0..7} U {16..23}
//   lanes16-31 : M=lane-16, K in {8..15} U {24..31}
// Each lane: two contiguous 8-half (16B) chunks -> 2x b128 loads.
static __device__ __forceinline__ half16 load_frag(const _Float16* base, int ld) {
    int lane = threadIdx.x & 31;
    int row  = lane & 15;
    int hh   = lane >> 4;
    const _Float16* p = base + row * ld + hh * 8;
    half8 lo = *(const half8*)(p);
    half8 hi = *(const half8*)(p + 16);
    half16 r;
#pragma unroll
    for (int i = 0; i < 8; ++i) { r[i] = lo[i]; r[i + 8] = hi[i]; }
    return r;
}

// ---------------------------------------------------------------------------
// Kernel 0: f32 -> f16 conversion (grid-stride)
// ---------------------------------------------------------------------------
__global__ __launch_bounds__(256) void k_convert(const float* __restrict__ src,
                                                 _Float16* __restrict__ dst, int n) {
    for (int i = blockIdx.x * blockDim.x + threadIdx.x; i < n;
         i += gridDim.x * blockDim.x)
        dst[i] = (_Float16)src[i];
}

// ---------------------------------------------------------------------------
// Kernel 1: QKV projection + bias + RoPE (Q,K) / transpose (V)
//   pid = blockIdx.y : 0=Q, 1=K, 2=V
//   out = x @ W^T + b ; each wave computes a 16x64 tile (one head's d_k)
// ---------------------------------------------------------------------------
__global__ __launch_bounds__(256) void k_qkv(const _Float16* __restrict__ xh,
                                             const _Float16* __restrict__ wh,
                                             const float* __restrict__ bq,
                                             const float* __restrict__ bk,
                                             const float* __restrict__ bv,
                                             _Float16* __restrict__ Qh,
                                             _Float16* __restrict__ Kh,
                                             _Float16* __restrict__ Vt) {
    const int A = BATCH * SEQ * DMODEL;      // 4194304
    const int W = DMODEL * DMODEL;           // 1048576
    int pid = blockIdx.y;
    const _Float16* x = xh + pid * A;
    const _Float16* w = wh + pid * W;
    const float* bias = (pid == 0) ? bq : (pid == 1) ? bk : bv;

    int wid  = blockIdx.x * 8 + (threadIdx.x >> 5);  // 0..4095
    int mt   = wid >> 4;                              // 0..255 (16-row tiles)
    int head = wid & 15;                              // 0..15  (64-col tiles)
    int rowbase = mt * 16;
    int nbase   = head * DK;

    float8 C[4];
#pragma unroll
    for (int j = 0; j < 4; ++j) C[j] = (float8)0.0f;

    for (int k0 = 0; k0 < DMODEL; k0 += 32) {
        half16 a = load_frag(x + rowbase * DMODEL + k0, DMODEL);
#pragma unroll
        for (int j = 0; j < 4; ++j) {
            half16 bfr = load_frag(w + (nbase + 16 * j) * DMODEL + k0, DMODEL);
            C[j] = wmma_f16(a, bfr, C[j]);
        }
    }

    int lane = threadIdx.x & 31;
    int n0   = lane & 15;
    int hh   = lane >> 4;

#pragma unroll
    for (int j = 0; j < 4; ++j) {
        float bj = bias[nbase + 16 * j + n0];
#pragma unroll
        for (int r = 0; r < 8; ++r) C[j][r] += bj;
    }

    if (pid == 2) {
        // V: store transposed Vt[b, h, d, s]
#pragma unroll
        for (int j = 0; j < 4; ++j) {
            int d = 16 * j + n0;
#pragma unroll
            for (int r = 0; r < 8; ++r) {
                int m = rowbase + r + 8 * hh;
                int b = m >> 11, s = m & (SEQ - 1);
                Vt[(((b * NHEAD + head) * DK + d) * SEQ) + s] = (_Float16)C[j][r];
            }
        }
    } else {
        // RoPE: angle(s,d) = s * 10000^(-(d%32)/32); rotate_half pairs d <-> d+-32
        float R[4][8];
#pragma unroll
        for (int j = 0; j < 4; ++j)
#pragma unroll
            for (int r = 0; r < 8; ++r)
                R[j][r] = (j < 2) ? -C[j + 2][r] : C[j - 2][r];

        _Float16* out = (pid == 0) ? Qh : Kh;
#pragma unroll
        for (int j = 0; j < 4; ++j) {
            int d = 16 * j + n0;
            float invf = __expf(-(float)(d & 31) * (9.210340371976184f / 32.0f));
#pragma unroll
            for (int r = 0; r < 8; ++r) {
                int m = rowbase + r + 8 * hh;
                int b = m >> 11, s = m & (SEQ - 1);
                float ang = (float)s * invf;
                float cs, sn;
                sincosf(ang, &sn, &cs);
                float v = C[j][r] * cs + R[j][r] * sn;
                out[(((b * NHEAD + head) * SEQ + s) * DK) + d] = (_Float16)v;
            }
        }
    }
}

// ---------------------------------------------------------------------------
// Kernel 2: flash attention. One wave = 16 query rows of one (b,h).
//   Streams 32-key blocks: S = Q K^T / 8, online softmax, O += P V.
// ---------------------------------------------------------------------------
__global__ __launch_bounds__(256) void k_attn(const _Float16* __restrict__ Qh,
                                              const _Float16* __restrict__ Kh,
                                              const _Float16* __restrict__ Vt,
                                              _Float16* __restrict__ AT) {
    __shared__ _Float16 Pl[8][16][32];   // per-wave P re-layout tile (8 KB)

    int wslot = threadIdx.x >> 5;
    int wid   = blockIdx.x * 8 + wslot;           // 0..4095
    int qt    = wid & 127;                         // 128 q-tiles per (b,h)
    int bh    = wid >> 7;                          // 0..31
    int b = bh >> 4, h = bh & 15;
    int qrow = qt * 16;

    int lane = threadIdx.x & 31;
    int n0 = lane & 15, hh = lane >> 4;

    const _Float16* qbase = Qh + ((size_t)(b * NHEAD + h) * SEQ + qrow) * DK;
    const _Float16* kbase = Kh + (size_t)(b * NHEAD + h) * SEQ * DK;
    const _Float16* vbase = Vt + (size_t)(b * NHEAD + h) * DK * SEQ;

    half16 Qf0 = load_frag(qbase + 0, DK);
    half16 Qf1 = load_frag(qbase + 32, DK);

    float8 O[4];
#pragma unroll
    for (int j = 0; j < 4; ++j) O[j] = (float8)0.0f;
    float m_i[8], l_i[8];
#pragma unroll
    for (int r = 0; r < 8; ++r) { m_i[r] = -3.0e38f; l_i[r] = 0.0f; }

    for (int kb = 0; kb < SEQ / 32; ++kb) {
        int kk0 = kb * 32;
        // scores: two 16-col tiles x two 32-wide d chunks
        float8 S0 = (float8)0.0f, S1 = (float8)0.0f;
        S0 = wmma_f16(Qf0, load_frag(kbase + (kk0 + 0) * DK + 0,  DK), S0);
        S0 = wmma_f16(Qf1, load_frag(kbase + (kk0 + 0) * DK + 32, DK), S0);
        S1 = wmma_f16(Qf0, load_frag(kbase + (kk0 + 16) * DK + 0,  DK), S1);
        S1 = wmma_f16(Qf1, load_frag(kbase + (kk0 + 16) * DK + 32, DK), S1);

        float rm[8], mn[8];
#pragma unroll
        for (int r = 0; r < 8; ++r) {
            S0[r] *= 0.125f; S1[r] *= 0.125f;
            rm[r] = fmaxf(S0[r], S1[r]);
        }
        // row max across the 16-lane half (row = VGPR idx + 8*half)
#pragma unroll
        for (int off = 1; off < 16; off <<= 1)
#pragma unroll
            for (int r = 0; r < 8; ++r)
                rm[r] = fmaxf(rm[r], __shfl_xor(rm[r], off, 16));

        float fac[8];
#pragma unroll
        for (int r = 0; r < 8; ++r) {
            mn[r]  = fmaxf(m_i[r], rm[r]);
            fac[r] = __expf(m_i[r] - mn[r]);
            m_i[r] = mn[r];
            l_i[r] *= fac[r];
        }
#pragma unroll
        for (int j = 0; j < 4; ++j)
#pragma unroll
            for (int r = 0; r < 8; ++r) O[j][r] *= fac[r];

        float rs[8];
#pragma unroll
        for (int r = 0; r < 8; ++r) {
            S0[r] = __expf(S0[r] - mn[r]);
            S1[r] = __expf(S1[r] - mn[r]);
            rs[r] = S0[r] + S1[r];
        }
#pragma unroll
        for (int off = 1; off < 16; off <<= 1)
#pragma unroll
            for (int r = 0; r < 8; ++r)
                rs[r] += __shfl_xor(rs[r], off, 16);
#pragma unroll
        for (int r = 0; r < 8; ++r) l_i[r] += rs[r];

        // C-fragment -> A-fragment re-layout via wave-private LDS
#pragma unroll
        for (int r = 0; r < 8; ++r) {
            Pl[wslot][r + 8 * hh][n0]      = (_Float16)S0[r];
            Pl[wslot][r + 8 * hh][16 + n0] = (_Float16)S1[r];
        }
        half16 Pf = load_frag(&Pl[wslot][0][0], 32);

#pragma unroll
        for (int dt = 0; dt < 4; ++dt)
            O[dt] = wmma_f16(Pf, load_frag(vbase + (dt * 16) * SEQ + kk0, SEQ), O[dt]);
    }

    // finalize: O /= l, write attn output as f16 [b, s, h*dk + d]
    float inv[8];
#pragma unroll
    for (int r = 0; r < 8; ++r) inv[r] = 1.0f / l_i[r];
#pragma unroll
    for (int dt = 0; dt < 4; ++dt) {
        int d = dt * 16 + n0;
#pragma unroll
        for (int r = 0; r < 8; ++r) {
            int s = qrow + r + 8 * hh;
            AT[((size_t)(b * SEQ + s) * DMODEL) + h * DK + d] =
                (_Float16)(O[dt][r] * inv[r]);
        }
    }
}

// ---------------------------------------------------------------------------
// Kernel 3: output projection  y = attn @ Wo^T + bo  (f32 out)
// ---------------------------------------------------------------------------
__global__ __launch_bounds__(256) void k_oproj(const _Float16* __restrict__ AT,
                                               const _Float16* __restrict__ Wo,
                                               const float* __restrict__ bo,
                                               float* __restrict__ out) {
    int wid = blockIdx.x * 8 + (threadIdx.x >> 5);
    int mt = wid >> 4;
    int nt = wid & 15;
    int rowbase = mt * 16;
    int nbase = nt * 64;

    float8 C[4];
#pragma unroll
    for (int j = 0; j < 4; ++j) C[j] = (float8)0.0f;

    for (int k0 = 0; k0 < DMODEL; k0 += 32) {
        half16 a = load_frag(AT + (size_t)rowbase * DMODEL + k0, DMODEL);
#pragma unroll
        for (int j = 0; j < 4; ++j) {
            half16 bfr = load_frag(Wo + (size_t)(nbase + 16 * j) * DMODEL + k0, DMODEL);
            C[j] = wmma_f16(a, bfr, C[j]);
        }
    }

    int lane = threadIdx.x & 31;
    int n0 = lane & 15, hh = lane >> 4;
#pragma unroll
    for (int j = 0; j < 4; ++j) {
        int col = nbase + 16 * j + n0;
        float bj = bo[col];
#pragma unroll
        for (int r = 0; r < 8; ++r) {
            int m = rowbase + r + 8 * hh;
            out[(size_t)m * DMODEL + col] = C[j][r] + bj;
        }
    }
}

// ---------------------------------------------------------------------------
extern "C" void kernel_launch(void* const* d_in, const int* in_sizes, int n_in,
                              void* d_out, int out_size, void* d_ws, size_t ws_size,
                              hipStream_t stream) {
    const float* query = (const float*)d_in[0];
    const float* key   = (const float*)d_in[1];
    const float* value = (const float*)d_in[2];
    const float* Wq = (const float*)d_in[3];
    const float* bq = (const float*)d_in[4];
    const float* Wk = (const float*)d_in[5];
    const float* bk = (const float*)d_in[6];
    const float* Wv = (const float*)d_in[7];
    const float* bv = (const float*)d_in[8];
    const float* Wo = (const float*)d_in[9];
    const float* bo = (const float*)d_in[10];
    float* out = (float*)d_out;

    const int A = BATCH * SEQ * DMODEL;   // 4194304
    const int W = DMODEL * DMODEL;        // 1048576

    _Float16* ws = (_Float16*)d_ws;
    _Float16* xh = ws;                    // 3*A : query/key/value f16
    _Float16* wh = ws + 3 * (size_t)A;    // 4*W : Wq/Wk/Wv/Wo f16
    _Float16* Qh = wh + 4 * (size_t)W;    // A : [b,h,s,d] (RoPE'd)
    _Float16* Kh = Qh + (size_t)A;        // A : [b,h,s,d] (RoPE'd)
    _Float16* Vt = Kh + (size_t)A;        // A : [b,h,d,s]
    _Float16* AT = Vt + (size_t)A;        // A : [b,s,D] attention output

    // 0) f32 -> f16 conversions
    k_convert<<<2048, 256, 0, stream>>>(query, xh + 0 * (size_t)A, A);
    k_convert<<<2048, 256, 0, stream>>>(key,   xh + 1 * (size_t)A, A);
    k_convert<<<2048, 256, 0, stream>>>(value, xh + 2 * (size_t)A, A);
    k_convert<<<1024, 256, 0, stream>>>(Wq, wh + 0 * (size_t)W, W);
    k_convert<<<1024, 256, 0, stream>>>(Wk, wh + 1 * (size_t)W, W);
    k_convert<<<1024, 256, 0, stream>>>(Wv, wh + 2 * (size_t)W, W);
    k_convert<<<1024, 256, 0, stream>>>(Wo, wh + 3 * (size_t)W, W);

    // 1) QKV projections + RoPE / V transpose
    dim3 g1(512, 3, 1);
    k_qkv<<<g1, 256, 0, stream>>>(xh, wh, bq, bk, bv, Qh, Kh, Vt);

    // 2) flash attention
    k_attn<<<512, 256, 0, stream>>>(Qh, Kh, Vt, AT);

    // 3) output projection
    k_oproj<<<512, 256, 0, stream>>>(AT, wh + 3 * (size_t)W, bo, out);
}
